// SchNetFeature_30322469110121
// MI455X (gfx1250) — matrix operand order repaired and checked by reading, not compiled, not run on hardware
//
#include <hip/hip_runtime.h>
#include <hip/hip_bf16.h>
#include <cstdint>
#include <cstddef>

// ---------------------------------------------------------------------------
// SchNet feature stack on gfx1250 (MI455X): filter-generating network runs on
// WMMA f16 16x16x32 with pairs striped across lanes; everything else is tiny
// per-atom VALU work. 3 interaction blocks need grid-wide sync on x, so the
// pipeline is a fixed sequence of kernel launches on `stream`.
// ---------------------------------------------------------------------------

typedef __attribute__((ext_vector_type(16))) _Float16    v16h;
typedef __attribute__((ext_vector_type(8)))  float       v8f;
typedef __attribute__((ext_vector_type(2)))  __fp16      v2hf;   // cvt_pkrtz result type
typedef __attribute__((ext_vector_type(4)))  float       v4f;
typedef __attribute__((ext_vector_type(4)))  unsigned int v4u;

#define NB     4
#define NATOM  512
#define FDIM   64
#define NRBF   25
#define KPAD   32

#define LOG2E     1.44269504088896340736f
#define LN2       0.69314718055994530942f
#define NGAMLOG2E (-10.0f * LOG2E)   // -gamma * log2(e) for base-2 RBF exponential

union U16H { v16h v; v4u q[2]; v2hf h[8]; };

__device__ __forceinline__ float ssp_f(float x) {
  // shifted softplus: softplus(x) - ln2, in base-2 form with raw hardware
  // transcendentals. log2 argument is in [1,2] so no denorm/inf fixups needed:
  //   ssp(x) = max(x,0) + ln2*log2(1 + 2^(-|x|*log2e)) - ln2
  float e = __builtin_amdgcn_exp2f(-fabsf(x) * LOG2E);
  float l = __builtin_amdgcn_logf(1.0f + e);          // v_log_f32 = log2
  return fmaf(LN2, l, fmaxf(x, 0.f) - LN2);
}

// ---------------------------------------------------------------------------
// Pack filter-net weights into transposed f16 operand arrays in workspace:
//   wA1[l][g][r] = fw1[l][r][g]  (r padded 25->32 with zeros)   [3][64][32]
//   wA2[l][f][g] = fw2[l][g][f]                                  [3][64][64]
// ---------------------------------------------------------------------------
__global__ void schnet_pack_weights(const float* __restrict__ fw1,
                                    const float* __restrict__ fw2,
                                    _Float16* __restrict__ wA1,
                                    _Float16* __restrict__ wA2) {
  const int n1 = 3 * FDIM * KPAD;
  const int n2 = 3 * FDIM * FDIM;
  for (int t = blockIdx.x * blockDim.x + threadIdx.x; t < n1 + n2;
       t += gridDim.x * blockDim.x) {
    if (t < n1) {
      int l = t / (FDIM * KPAD);
      int g = (t / KPAD) % FDIM;
      int r = t % KPAD;
      float val = (r < NRBF) ? fw1[(l * NRBF + r) * FDIM + g] : 0.f;
      wA1[t] = (_Float16)val;
    } else {
      int u = t - n1;
      int l = u / (FDIM * FDIM);
      int f = (u / FDIM) % FDIM;
      int g = u % FDIM;
      wA2[u] = (_Float16)fw2[(l * FDIM + g) * FDIM + f];
    }
  }
}

// ---------------------------------------------------------------------------
// f = x @ in2f[layer]   (per-atom 64x64 linear, no bias)
// ---------------------------------------------------------------------------
__global__ __launch_bounds__(64)
void schnet_atomwise_in(const float* __restrict__ x,
                        const float* __restrict__ in2f,
                        float* __restrict__ fmat, int layer) {
  __shared__ float xr[FDIM];
  const int a = blockIdx.x;
  const int f = threadIdx.x;
  xr[f] = x[(size_t)a * FDIM + f];
  __syncthreads();
  const float* w = in2f + (size_t)layer * FDIM * FDIM;
  float acc = 0.f;
#pragma unroll 8
  for (int k = 0; k < FDIM; ++k) acc = fmaf(xr[k], w[k * FDIM + f], acc);
  fmat[(size_t)a * FDIM + f] = acc;
}

// ---------------------------------------------------------------------------
// Continuous-filter convolution. One wave per target atom (b,i).
// Per 16-j tile:  W1^T = fw1^T x rbf^T   (4 WMMA, K=32)
//                 ssp(+fb1) -> f16 -> LDS (ds_store_b128)
//                 W^T  = fw2^T x W1^T    (8 WMMA, K=64 as 2x32)
//                 yacc[f] += (W^T + fb2) * f_j[f]
// Final: shfl_xor reduce over the 16 pair-lanes of each half, store y.
// ---------------------------------------------------------------------------
__global__ __launch_bounds__(32)
void schnet_cfconv(const float* __restrict__ distances,
                   const float* __restrict__ fmat,
                   const _Float16* __restrict__ wA1,
                   const _Float16* __restrict__ wA2,
                   const float* __restrict__ fb1,
                   const float* __restrict__ fb2,
                   float* __restrict__ yout, int layer) {
  __shared__ __align__(16) _Float16 ldsA1[FDIM * KPAD];   // 4 KB
  __shared__ __align__(16) _Float16 ldsA2[FDIM * FDIM];   // 8 KB
  __shared__ __align__(16) _Float16 ldsW1[16 * FDIM];     // 2 KB

  const int lane = threadIdx.x;
  const int half = lane >> 4;   // 0: lanes 0-15, 1: lanes 16-31
  const int mrow = lane & 15;   // M-row / pair column index
  const int b = blockIdx.x >> 9;
  const int i = blockIdx.x & (NATOM - 1);

  // Stage packed filter weights into LDS (b128 copies).
  {
    const v4u* s1 = reinterpret_cast<const v4u*>(wA1 + (size_t)layer * FDIM * KPAD);
    v4u* d1 = reinterpret_cast<v4u*>(ldsA1);
#pragma unroll
    for (int t = 0; t < 8; ++t) d1[lane + 32 * t] = s1[lane + 32 * t];      // 4 KB
    const v4u* s2 = reinterpret_cast<const v4u*>(wA2 + (size_t)layer * FDIM * FDIM);
    v4u* d2 = reinterpret_cast<v4u*>(ldsA2);
#pragma unroll
    for (int t = 0; t < 16; ++t) d2[lane + 32 * t] = s2[lane + 32 * t];     // 8 KB
  }
  __syncthreads();

  // A operands resident in registers, contiguous k order (same as B):
  // lane holds row M = mrow; K bytes [0..31] (half 0) or
  // [32..63] (half 1) of the 32-element f16 row.
  U16H A1[4];
#pragma unroll
  for (int t = 0; t < 4; ++t) {
    const v4u* p = reinterpret_cast<const v4u*>(ldsA1 + (t * 16 + mrow) * KPAD);
    A1[t].q[0] = p[2 * half];
    A1[t].q[1] = p[2 * half + 1];
  }
  U16H A2[4][2];
#pragma unroll
  for (int ft = 0; ft < 4; ++ft) {
    const v4u* p = reinterpret_cast<const v4u*>(ldsA2 + (ft * 16 + mrow) * FDIM);
#pragma unroll
    for (int c = 0; c < 2; ++c) {
      A2[ft][c].q[0] = p[4 * c + 2 * half];
      A2[ft][c].q[1] = p[4 * c + 2 * half + 1];
    }
  }

  // Biases mapped to the C/D layout: VGPR v of tile t holds row g/f = 16t+8*half+v.
  float b1v[4][8], b2v[4][8];
#pragma unroll
  for (int t = 0; t < 4; ++t)
#pragma unroll
    for (int v = 0; v < 8; ++v) {
      b1v[t][v] = fb1[layer * FDIM + t * 16 + half * 8 + v];
      b2v[t][v] = fb2[layer * FDIM + t * 16 + half * 8 + v];
    }

  float yacc[4][8];
#pragma unroll
  for (int t = 0; t < 4; ++t)
#pragma unroll
    for (int v = 0; v < 8; ++v) yacc[t][v] = 0.f;

  const v8f zero8 = {0.f, 0.f, 0.f, 0.f, 0.f, 0.f, 0.f, 0.f};
  const float* drow = distances + ((size_t)(b * NATOM + i)) * NATOM;

  for (int jt = 0; jt < NATOM / 16; ++jt) {
    const int j = jt * 16 + mrow;
    const float dist = drow[j];
    if (jt + 1 < NATOM / 16) __builtin_prefetch(&drow[j + 16], 0, 0);

    // B operand of GEMM1: rbf^T tile (K = rbf index, N = pair).
    // Lane supplies column `mrow`, K slots 16*half .. 16*half+15.
    // rbf = exp(-10*(d-mu)^2) = 2^(t*t * (-10*log2e)) -- raw v_exp_f32.
    U16H Brbf;
    const int rbase = half * 16;
#pragma unroll
    for (int v = 0; v < 8; ++v) {
      const int r0 = rbase + 2 * v, r1 = r0 + 1;
      float t0 = dist - (float)r0 * (5.0f / 24.0f);
      float t1 = dist - (float)r1 * (5.0f / 24.0f);
      float g0 = __builtin_amdgcn_exp2f(t0 * t0 * NGAMLOG2E);
      float g1 = __builtin_amdgcn_exp2f(t1 * t1 * NGAMLOG2E);
      g0 = (r0 < NRBF) ? g0 : 0.f;   // zero-pad K 25..31
      g1 = (r1 < NRBF) ? g1 : 0.f;
      Brbf.h[v] = __builtin_amdgcn_cvt_pkrtz(g0, g1);
    }

    // GEMM1 + bias + ssp, packed f16 into LDS as W1^T stored [pair][g].
#pragma unroll
    for (int t = 0; t < 4; ++t) {
      v8f d1 = __builtin_amdgcn_wmma_f32_16x16x32_f16(
          false, A1[t].v, false, Brbf.v, (short)0, zero8, false, false);
      U16H pk;
#pragma unroll
      for (int v = 0; v < 4; ++v) {
        float w0 = ssp_f(d1[2 * v] + b1v[t][2 * v]);
        float w1 = ssp_f(d1[2 * v + 1] + b1v[t][2 * v + 1]);
        pk.h[v] = __builtin_amdgcn_cvt_pkrtz(w0, w1);
      }
      *reinterpret_cast<v4u*>(ldsW1 + mrow * FDIM + t * 16 + half * 8) = pk.q[0];
    }
    __syncthreads();

    // B operands of GEMM2 from LDS: lane = pair column, K = g chunk.
    U16H Bw[2];
#pragma unroll
    for (int c = 0; c < 2; ++c) {
      const v4u* p = reinterpret_cast<const v4u*>(ldsW1 + mrow * FDIM + c * 32 + half * 16);
      Bw[c].q[0] = p[0];
      Bw[c].q[1] = p[1];
    }

    // GEMM2 (K=64 as two 32-chunks) + bias, then conv FMA with f_j[f].
    const float* fj = fmat + ((size_t)(b * NATOM + j)) * FDIM;
#pragma unroll
    for (int ft = 0; ft < 4; ++ft) {
      v8f acc = __builtin_amdgcn_wmma_f32_16x16x32_f16(
          false, A2[ft][0].v, false, Bw[0].v, (short)0, zero8, false, false);
      acc = __builtin_amdgcn_wmma_f32_16x16x32_f16(
          false, A2[ft][1].v, false, Bw[1].v, (short)0, acc, false, false);
      const v4f f0 = *reinterpret_cast<const v4f*>(fj + ft * 16 + half * 8);
      const v4f f1 = *reinterpret_cast<const v4f*>(fj + ft * 16 + half * 8 + 4);
#pragma unroll
      for (int v = 0; v < 4; ++v) {
        yacc[ft][v]     = fmaf(acc[v]     + b2v[ft][v],     f0[v], yacc[ft][v]);
        yacc[ft][v + 4] = fmaf(acc[v + 4] + b2v[ft][v + 4], f1[v], yacc[ft][v + 4]);
      }
    }
    __syncthreads();
  }

  // Sum over the 16 pair-lanes inside each half (xor over lane bits 0..3).
#pragma unroll
  for (int m = 1; m < 16; m <<= 1)
#pragma unroll
    for (int ft = 0; ft < 4; ++ft)
#pragma unroll
      for (int v = 0; v < 8; ++v)
        yacc[ft][v] += __shfl_xor(yacc[ft][v], m, 32);

  if (mrow == 0) {
    float* yo = yout + ((size_t)(b * NATOM + i)) * FDIM + half * 8;
#pragma unroll
    for (int ft = 0; ft < 4; ++ft)
#pragma unroll
      for (int v = 0; v < 8; ++v) yo[ft * 16 + v] = yacc[ft][v];
  }
}

// ---------------------------------------------------------------------------
// x_out = x_in + (ssp(y @ f2o + f2b)) @ dw + db
// ---------------------------------------------------------------------------
__global__ __launch_bounds__(64)
void schnet_post(const float* __restrict__ y,
                 const float* __restrict__ f2o, const float* __restrict__ f2b,
                 const float* __restrict__ dw, const float* __restrict__ db,
                 const float* __restrict__ xin, float* __restrict__ xout,
                 int layer) {
  __shared__ float yr[FDIM];
  __shared__ float t1[FDIM];
  const int a = blockIdx.x;
  const int f = threadIdx.x;
  yr[f] = y[(size_t)a * FDIM + f];
  __syncthreads();
  const float* w1 = f2o + (size_t)layer * FDIM * FDIM;
  float acc = f2b[layer * FDIM + f];
#pragma unroll 8
  for (int k = 0; k < FDIM; ++k) acc = fmaf(yr[k], w1[k * FDIM + f], acc);
  t1[f] = ssp_f(acc);
  __syncthreads();
  const float* w2 = dw + (size_t)layer * FDIM * FDIM;
  float acc2 = db[layer * FDIM + f];
#pragma unroll 8
  for (int k = 0; k < FDIM; ++k) acc2 = fmaf(t1[k], w2[k * FDIM + f], acc2);
  xout[(size_t)a * FDIM + f] = xin[(size_t)a * FDIM + f] + acc2;
}

// ---------------------------------------------------------------------------
// out = ssp(x @ aw1 + ab1) @ aw2 + ab2
// ---------------------------------------------------------------------------
__global__ __launch_bounds__(64)
void schnet_readout(const float* __restrict__ x,
                    const float* __restrict__ aw1, const float* __restrict__ ab1,
                    const float* __restrict__ aw2, const float* __restrict__ ab2,
                    float* __restrict__ out) {
  __shared__ float xr[FDIM];
  __shared__ float hs[32];
  const int a = blockIdx.x;
  const int t = threadIdx.x;
  xr[t] = x[(size_t)a * FDIM + t];
  __syncthreads();
  if (t < 32) {
    float acc = ab1[t];
#pragma unroll 8
    for (int k = 0; k < FDIM; ++k) acc = fmaf(xr[k], aw1[k * 32 + t], acc);
    hs[t] = ssp_f(acc) * aw2[t];
  }
  __syncthreads();
  if (t == 0) {
    float s = ab2[0];
    for (int k = 0; k < 32; ++k) s += hs[k];
    out[a] = s;
  }
}

// ---------------------------------------------------------------------------
extern "C" void kernel_launch(void* const* d_in, const int* in_sizes, int n_in,
                              void* d_out, int out_size, void* d_ws, size_t ws_size,
                              hipStream_t stream) {
  (void)in_sizes; (void)n_in; (void)out_size; (void)ws_size;

  const float* embedding = (const float*)d_in[0];
  const float* distances = (const float*)d_in[1];
  const float* in2f = (const float*)d_in[2];
  const float* fw1  = (const float*)d_in[3];
  const float* fb1  = (const float*)d_in[4];
  const float* fw2  = (const float*)d_in[5];
  const float* fb2  = (const float*)d_in[6];
  const float* f2o  = (const float*)d_in[7];
  const float* f2b  = (const float*)d_in[8];
  const float* dw   = (const float*)d_in[9];
  const float* db   = (const float*)d_in[10];
  const float* aw1  = (const float*)d_in[11];
  const float* ab1  = (const float*)d_in[12];
  const float* aw2  = (const float*)d_in[13];
  const float* ab2  = (const float*)d_in[14];
  float* out = (float*)d_out;

  // Workspace layout (all 16B aligned):
  char* ws = (char*)d_ws;
  _Float16* wA1 = (_Float16*)(ws);                 // 3*64*32 f16 = 12288 B
  _Float16* wA2 = (_Float16*)(ws + 12288);         // 3*64*64 f16 = 24576 B
  float* fmat = (float*)(ws + 36864);              // 2048*64 f32 = 512 KB
  float* ybuf = (float*)(ws + 36864 + 524288);     // 512 KB
  float* xbuf = (float*)(ws + 36864 + 2 * 524288); // 512 KB

  const int atoms = NB * NATOM;  // 2048

  schnet_pack_weights<<<32, 256, 0, stream>>>(fw1, fw2, wA1, wA2);
  for (int layer = 0; layer < 3; ++layer) {
    const float* xsrc = (layer == 0) ? embedding : xbuf;
    schnet_atomwise_in<<<atoms, 64, 0, stream>>>(xsrc, in2f, fmat, layer);
    schnet_cfconv<<<atoms, 32, 0, stream>>>(distances, fmat, wA1, wA2,
                                            fb1, fb2, ybuf, layer);
    schnet_post<<<atoms, 64, 0, stream>>>(ybuf, f2o, f2b, dw, db,
                                          xsrc, xbuf, layer);
  }
  schnet_readout<<<atoms, 64, 0, stream>>>(xbuf, aw1, ab1, aw2, ab2, out);
}
